// RADFE_79465484910741
// MI455X (gfx1250) — matrix operand order repaired and checked by reading, not compiled, not run on hardware
//
#include <hip/hip_runtime.h>

// ============================================================================
// MI455X / gfx1250 implementation of the radar Mamba pipeline.
// Dense projections: V_WMMA_F32_16X16X32_F16, 8-wave blocks, 128x64 panels,
// f16 weight panel staged in LDS (shared across waves), 4 WMMA per K-step.
// Sequential selective scan is scalar f32 with LDS-staged B/C rows.
// ============================================================================

typedef __attribute__((ext_vector_type(16))) _Float16 v16h;
typedef __attribute__((ext_vector_type(8)))  float    v8f;

static __device__ __forceinline__ float dsigmoid(float v) {
    return 1.0f / (1.0f + __expf(-v));
}
static __device__ __forceinline__ float dsilu(float v) {
    return v * dsigmoid(v);
}

// ---------------------------------------------------------------------------
// WMMA GEMM:  C[m,n] = act( sum_k A[m,k] * W[n,k] + bias[n] )
//   A: M x K row-major (stride lda), W: N x K row-major, C: M x N (stride ldc)
//   act: 0 = none, 1 = SiLU.
// Block = 256 threads = 8 waves. Block tile = 128 (M) x 64 (N).
//   - 64 x K weight panel converted to f16 into LDS once per block,
//     zero-padded in N so the WMMA loop is branch-free (EXEC all ones).
//   - Each wave owns one 16-row m-tile; per K-step: 1 A fragment (global),
//     4 B fragments (LDS), 4 x v_wmma_f32_16x16x32_f16.
// Fragment layouts used here (wave32; A/B k order contiguous, C/D per CDNA5 ISA 7.12.2):
//   A (16x32 f16): row = lane&15; g=lane>>4; el i -> K = 16g + i
//   B (32x16 f16): col = lane&15; el i -> K = 16g + i
//   C/D (16x16 f32): el e -> M = 8g+e, N = lane&15
// Requires: M % 128 == 0, K % 32 == 0, K <= 512.
// ---------------------------------------------------------------------------
__global__ __launch_bounds__(256)
void k_gemm_wmma(const float* __restrict__ A, const float* __restrict__ W,
                 const float* __restrict__ bias, float* __restrict__ C,
                 int M, int N, int K, int lda, int ldc, int act)
{
    __shared__ _Float16 Wl[64 * 512];   // 64 KB: 64-row x K f16 weight panel

    const int tid   = threadIdx.x;
    const int nbase = blockIdx.y * 64;

    // ---- cooperative stage: W panel -> LDS (f16, zero-pad rows >= N) ----
    for (int n = tid >> 6; n < 64; n += 4) {          // 4 rows in flight, 64 lanes/row
        const int ng = nbase + n;
        if (ng < N) {
            const float* wrow = W + (size_t)ng * K;
            for (int k = (tid & 63); k < K; k += 64)
                Wl[n * K + k] = (_Float16)wrow[k];
        } else {
            for (int k = (tid & 63); k < K; k += 64)
                Wl[n * K + k] = (_Float16)0.0f;
        }
    }
    __syncthreads();

    const int wave = tid >> 5;
    const int lane = tid & 31;
    const int g = lane >> 4;
    const int r = lane & 15;
    const int mrow = blockIdx.x * 128 + wave * 16 + r;   // A row for this lane

    v8f acc[4] = {};
    const float* arow = A + (size_t)mrow * lda;

    for (int kt = 0; kt < K; kt += 32) {
        __builtin_prefetch(arow + kt + 64, 0, 3);        // global_prefetch_b8
        v16h a;
        #pragma unroll
        for (int i = 0; i < 16; ++i) a[i] = (_Float16)arow[kt + 16 * g + i];

        const _Float16* wp = &Wl[r * K + kt + 16 * g];   // n-tile 0, this lane's column
        #pragma unroll
        for (int tn = 0; tn < 4; ++tn) {
            v16h b;
            const _Float16* wt = wp + tn * 16 * K;       // +16 LDS rows per n-tile
            #pragma unroll
            for (int i = 0; i < 16; ++i) b[i] = wt[i];
            acc[tn] = __builtin_amdgcn_wmma_f32_16x16x32_f16(
                /*neg_a=*/false, a, /*neg_b=*/false, b,
                /*c_mod=*/(short)0, acc[tn], /*reuse_a=*/false, /*reuse_b=*/false);
        }
    }

    // ---- epilogue: bias + activation + guarded store ----
    #pragma unroll
    for (int tn = 0; tn < 4; ++tn) {
        const int nn = nbase + tn * 16 + r;
        if (nn < N) {
            float bi = bias ? bias[nn] : 0.0f;
            #pragma unroll
            for (int e = 0; e < 8; ++e) {
                const int mm = blockIdx.x * 128 + wave * 16 + 8 * g + e;
                float v = acc[tn][e] + bi;
                if (act == 1) v = dsilu(v);
                C[(size_t)mm * ldc + nn] = v;
            }
        }
    }
}

// ---------------------------------------------------------------------------
// LayerNorm over last dim (one thread owns one row; safe in-place).
// ---------------------------------------------------------------------------
__global__ void k_layernorm(const float* __restrict__ in, float* __restrict__ out,
                            const float* __restrict__ w, const float* __restrict__ b,
                            int M, int Cdim)
{
    int row = blockIdx.x * blockDim.x + threadIdx.x;
    if (row >= M) return;
    const float* p = in + (size_t)row * Cdim;
    float m = 0.0f;
    for (int i = 0; i < Cdim; ++i) m += p[i];
    m /= (float)Cdim;
    float v = 0.0f;
    for (int i = 0; i < Cdim; ++i) { float d = p[i] - m; v += d * d; }
    v /= (float)Cdim;
    float inv = rsqrtf(v + 1e-5f);
    float* q = out + (size_t)row * Cdim;
    for (int i = 0; i < Cdim; ++i) q[i] = (p[i] - m) * inv * w[i] + b[i];
}

// (D0,D1,C) -> (D1,D0,C)
__global__ void k_transpose01(const float* __restrict__ in, float* __restrict__ out,
                              int D0, int D1, int C)
{
    size_t idx = (size_t)blockIdx.x * blockDim.x + threadIdx.x;
    size_t total = (size_t)D0 * D1 * C;
    if (idx >= total) return;
    int c = (int)(idx % C);
    size_t t = idx / C;
    int j = (int)(t % D1);
    int i = (int)(t / D1);
    out[((size_t)j * D0 + i) * C + c] = in[idx];
}

// (D0,D1,D2) -> (D1,D2,D0)
__global__ void k_permute120(const float* __restrict__ in, float* __restrict__ out,
                             int D0, int D1, int D2)
{
    size_t idx = (size_t)blockIdx.x * blockDim.x + threadIdx.x;
    size_t total = (size_t)D0 * D1 * D2;
    if (idx >= total) return;
    int d2 = (int)(idx % D2);
    size_t t = idx / D2;
    int d1 = (int)(t % D1);
    int d0 = (int)(t / D1);
    out[((size_t)d1 * D2 + d2) * D0 + d0] = in[idx];
}

// (D0,D1,D2) -> (D0,D2,D1)
__global__ void k_permute021(const float* __restrict__ in, float* __restrict__ out,
                             int D0, int D1, int D2)
{
    size_t idx = (size_t)blockIdx.x * blockDim.x + threadIdx.x;
    size_t total = (size_t)D0 * D1 * D2;
    if (idx >= total) return;
    int d2 = (int)(idx % D2);
    size_t t = idx / D2;
    int d1 = (int)(t % D1);
    int d0 = (int)(t / D1);
    out[((size_t)d0 * D2 + d2) * D1 + d1] = in[idx];
}

// Depthwise causal 1D conv (width 4) + bias + SiLU.
// xz rows = (b*L + t), channel d at column (off_in + d), row stride ld_in.
__global__ void k_dwconv_silu(const float* __restrict__ xz, int ld_in, int off_in,
                              const float* __restrict__ w, const float* __restrict__ wb,
                              float* __restrict__ u, int Bn, int L, int D)
{
    size_t idx = (size_t)blockIdx.x * blockDim.x + threadIdx.x;
    size_t total = (size_t)Bn * L * D;
    if (idx >= total) return;
    int d = (int)(idx % D);
    size_t t0 = idx / D;
    int t = (int)(t0 % L);
    int b = (int)(t0 / L);
    float s = 0.0f;
    #pragma unroll
    for (int j = 0; j < 4; ++j) {
        int tt = t + j - 3;
        if (tt >= 0)
            s += w[d * 4 + j] * xz[((size_t)b * L + tt) * ld_in + off_in + d];
    }
    s += wb[d];
    u[idx] = dsilu(s);
}

// dt = softplus( x_dbl[:, :R] @ dtw^T + dtb )
__global__ void k_dtproj(const float* __restrict__ xdbl, int ldx,
                         const float* __restrict__ dtw, const float* __restrict__ dtb,
                         float* __restrict__ dt, int Mrows, int D, int R)
{
    size_t idx = (size_t)blockIdx.x * blockDim.x + threadIdx.x;
    if (idx >= (size_t)Mrows * D) return;
    int d = (int)(idx % D);
    size_t m = idx / D;
    float s = dtb[d];
    for (int r = 0; r < R; ++r) s += xdbl[m * ldx + r] * dtw[d * R + r];
    dt[idx] = (s > 20.0f) ? s : log1pf(__expf(s));
}

// Selective scan: one block per sequence b, one thread per inner channel d.
// State h[32] lives in VGPRs; B/C rows staged through LDS each step.
__global__ __launch_bounds__(128)
void k_scan(const float* __restrict__ u, const float* __restrict__ dt,
            const float* __restrict__ Alog,
            const float* __restrict__ xdbl, int ldx, int offB, int offC,
            const float* __restrict__ Dp, float* __restrict__ y,
            int L, int Dh)
{
    const int b = blockIdx.x;
    const int d = threadIdx.x;
    __shared__ float Bs[32];
    __shared__ float Cs[32];

    float h[32];
    float Arow[32];
    #pragma unroll
    for (int n = 0; n < 32; ++n) {
        h[n] = 0.0f;
        Arow[n] = -__expf(Alog[d * 32 + n]);
    }
    const float Dd = Dp[d];

    for (int t = 0; t < L; ++t) {
        const size_t row = (size_t)b * L + t;
        if (d < 32)       Bs[d]      = xdbl[row * ldx + offB + d];
        else if (d < 64)  Cs[d - 32] = xdbl[row * ldx + offC + (d - 32)];
        __syncthreads();
        const float dtv = dt[row * Dh + d];
        const float uv  = u[row * Dh + d];
        float acc = 0.0f;
        #pragma unroll
        for (int n = 0; n < 32; ++n) {
            float dA = __expf(dtv * Arow[n]);
            h[n] = dA * h[n] + dtv * Bs[n] * uv;
            acc += h[n] * Cs[n];
        }
        y[row * Dh + d] = acc + uv * Dd;
        __syncthreads();
    }
}

// y *= silu(z), z taken from xz[:, offz + d] with row stride ld.
__global__ void k_gate(float* __restrict__ y, const float* __restrict__ xz,
                       int ld, int offz, int Mrows, int Dh)
{
    size_t idx = (size_t)blockIdx.x * blockDim.x + threadIdx.x;
    if (idx >= (size_t)Mrows * Dh) return;
    int d = (int)(idx % Dh);
    size_t m = idx / Dh;
    float z = xz[m * ld + offz + d];
    y[idx] *= dsilu(z);
}

// 1x1 conv over channels (NCHW with N=1): out[co,hw] = sum_ci w[co,ci]*in[ci,hw]+b
__global__ void k_conv1x1(const float* __restrict__ in, const float* __restrict__ w,
                          const float* __restrict__ b, float* __restrict__ out,
                          int Cin, int Cout, int HW)
{
    size_t idx = (size_t)blockIdx.x * blockDim.x + threadIdx.x;
    if (idx >= (size_t)Cout * HW) return;
    int hw = (int)(idx % HW);
    int co = (int)(idx / HW);
    float s = b ? b[co] : 0.0f;
    for (int ci = 0; ci < Cin; ++ci) s += w[co * Cin + ci] * in[(size_t)ci * HW + hw];
    out[idx] = s;
}

// 3x3 SAME conv, NCHW N=1.  act: 0 none, 1 silu, 2 sigmoid.
__global__ void k_conv3x3(const float* __restrict__ in, const float* __restrict__ w,
                          const float* __restrict__ b, float* __restrict__ out,
                          int Cin, int Cout, int H, int W, int act)
{
    size_t idx = (size_t)blockIdx.x * blockDim.x + threadIdx.x;
    if (idx >= (size_t)Cout * H * W) return;
    int x = (int)(idx % W);
    size_t t = idx / W;
    int y = (int)(t % H);
    int co = (int)(t / H);
    float s = b ? b[co] : 0.0f;
    for (int ci = 0; ci < Cin; ++ci) {
        #pragma unroll
        for (int ky = 0; ky < 3; ++ky) {
            int yy = y + ky - 1;
            if (yy < 0 || yy >= H) continue;
            #pragma unroll
            for (int kx = 0; kx < 3; ++kx) {
                int xx = x + kx - 1;
                if (xx < 0 || xx >= W) continue;
                s += w[((co * Cin + ci) * 3 + ky) * 3 + kx] * in[((size_t)ci * H + yy) * W + xx];
            }
        }
    }
    if (act == 1) s = dsilu(s);
    else if (act == 2) s = dsigmoid(s);
    out[idx] = s;
}

// Training-mode BatchNorm2d (N=1): per-channel batch stats over H*W, in-place.
__global__ __launch_bounds__(256)
void k_batchnorm(float* __restrict__ x, const float* __restrict__ g,
                 const float* __restrict__ b, int HW)
{
    const int c = blockIdx.x;
    __shared__ float ssum[256];
    __shared__ float ssq[256];
    float s = 0.0f, q = 0.0f;
    float* p = x + (size_t)c * HW;
    for (int i = threadIdx.x; i < HW; i += blockDim.x) {
        float v = p[i]; s += v; q += v * v;
    }
    ssum[threadIdx.x] = s; ssq[threadIdx.x] = q;
    __syncthreads();
    for (int st = 128; st > 0; st >>= 1) {
        if ((int)threadIdx.x < st) {
            ssum[threadIdx.x] += ssum[threadIdx.x + st];
            ssq[threadIdx.x]  += ssq[threadIdx.x + st];
        }
        __syncthreads();
    }
    float mean = ssum[0] / (float)HW;
    float var  = ssq[0] / (float)HW - mean * mean;
    float inv  = rsqrtf(var + 1e-5f) * g[c];
    float beta = b[c] - mean * inv;
    for (int i = threadIdx.x; i < HW; i += blockDim.x)
        p[i] = p[i] * inv + beta;
}

// Bilinear resize (half-pixel centers, edge clamp), NCHW N=1.
__global__ void k_resize(const float* __restrict__ in, float* __restrict__ out,
                         int C, int Hi, int Wi, int Ho, int Wo)
{
    size_t idx = (size_t)blockIdx.x * blockDim.x + threadIdx.x;
    if (idx >= (size_t)C * Ho * Wo) return;
    int x = (int)(idx % Wo);
    size_t t = idx / Wo;
    int y = (int)(t % Ho);
    int c = (int)(t / Ho);
    float fy = ((float)y + 0.5f) * (float)Hi / (float)Ho - 0.5f;
    float fx = ((float)x + 0.5f) * (float)Wi / (float)Wo - 0.5f;
    int y0 = (int)floorf(fy), x0 = (int)floorf(fx);
    float wy = fy - (float)y0, wx = fx - (float)x0;
    int y1 = y0 + 1, x1 = x0 + 1;
    y0 = min(max(y0, 0), Hi - 1); y1 = min(max(y1, 0), Hi - 1);
    x0 = min(max(x0, 0), Wi - 1); x1 = min(max(x1, 0), Wi - 1);
    const float* p = in + (size_t)c * Hi * Wi;
    float v00 = p[(size_t)y0 * Wi + x0], v01 = p[(size_t)y0 * Wi + x1];
    float v10 = p[(size_t)y1 * Wi + x0], v11 = p[(size_t)y1 * Wi + x1];
    float top = v00 + (v01 - v00) * wx;
    float bot = v10 + (v11 - v10) * wx;
    out[idx] = top + (bot - top) * wy;
}

// ============================================================================
// Host orchestration
// ============================================================================
extern "C" void kernel_launch(void* const* d_in, const int* in_sizes, int n_in,
                              void* d_out, int out_size, void* d_ws, size_t ws_size,
                              hipStream_t stream)
{
    (void)in_sizes; (void)n_in; (void)out_size; (void)ws_size;

    // --- inputs, in setup_inputs() dict (insertion) order, leaves flattened ---
    const float* x          = (const float*)d_in[0];
    const float* in_ln_w    = (const float*)d_in[1];
    const float* in_ln_b    = (const float*)d_in[2];
    const float* fast_lin_w = (const float*)d_in[3];
    const float* fast_lin_b = (const float*)d_in[4];
    const float* fast_ln_w  = (const float*)d_in[5];
    const float* fast_ln_b  = (const float*)d_in[6];
    // fast_ssm (d_model=32, d_inner=64, N=32, dt_rank=2)
    const float* f_in_proj  = (const float*)d_in[7];   // (128,32)
    const float* f_conv_w   = (const float*)d_in[8];   // (64,1,4)
    const float* f_conv_b   = (const float*)d_in[9];
    const float* f_x_proj   = (const float*)d_in[10];  // (66,64)
    const float* f_dtw      = (const float*)d_in[11];  // (64,2)
    const float* f_dtb      = (const float*)d_in[12];
    const float* f_Alog     = (const float*)d_in[13];  // (64,32)
    const float* f_D        = (const float*)d_in[14];
    const float* f_out_proj = (const float*)d_in[15];  // (32,64)
    const float* f_wrap_w   = (const float*)d_in[16];  // (32,32)
    const float* f_wrap_b   = (const float*)d_in[17];
    const float* slow_lin_w = (const float*)d_in[18];  // (64,32)
    const float* slow_lin_b = (const float*)d_in[19];
    const float* slow_ln_w  = (const float*)d_in[20];
    const float* slow_ln_b  = (const float*)d_in[21];
    // slow_ssm (d_model=64, d_inner=128, N=32, dt_rank=4)
    const float* s_in_proj  = (const float*)d_in[22];  // (256,64)
    const float* s_conv_w   = (const float*)d_in[23];  // (128,1,4)
    const float* s_conv_b   = (const float*)d_in[24];
    const float* s_x_proj   = (const float*)d_in[25];  // (68,128)
    const float* s_dtw      = (const float*)d_in[26];  // (128,4)
    const float* s_dtb      = (const float*)d_in[27];
    const float* s_Alog     = (const float*)d_in[28];  // (128,32)
    const float* s_D        = (const float*)d_in[29];
    const float* s_out_proj = (const float*)d_in[30];  // (64,128)
    const float* s_wrap_w   = (const float*)d_in[31];  // (64,64)
    const float* s_wrap_b   = (const float*)d_in[32];
    const float* range_w    = (const float*)d_in[33];  // (32,512)
    const float* range_b    = (const float*)d_in[34];
    const float* range_ln_w = (const float*)d_in[35];
    const float* range_ln_b = (const float*)d_in[36];
    const float* az_w       = (const float*)d_in[37];  // (56,64)
    const float* az_b       = (const float*)d_in[38];
    const float* az_ln_w    = (const float*)d_in[39];
    const float* az_ln_b    = (const float*)d_in[40];
    const float* pool_det_w = (const float*)d_in[41];  // (16,256,1,1)
    const float* pool_det_b = (const float*)d_in[42];
    const float* conv1_w    = (const float*)d_in[43];
    const float* bn1_g      = (const float*)d_in[44];
    const float* bn1_b      = (const float*)d_in[45];
    const float* conv2_w    = (const float*)d_in[46];
    const float* bn2_g      = (const float*)d_in[47];
    const float* bn2_b      = (const float*)d_in[48];
    const float* conv3_w    = (const float*)d_in[49];
    const float* bn3_g      = (const float*)d_in[50];
    const float* bn3_b      = (const float*)d_in[51];
    const float* conv4_w    = (const float*)d_in[52];
    const float* bn4_g      = (const float*)d_in[53];
    const float* bn4_b      = (const float*)d_in[54];
    const float* cls_w      = (const float*)d_in[55];
    const float* cls_b      = (const float*)d_in[56];
    const float* reg_w      = (const float*)d_in[57];
    const float* reg_b      = (const float*)d_in[58];
    const float* pool_seg_w = (const float*)d_in[59];
    const float* pool_seg_b = (const float*)d_in[60];
    const float* conv0_w    = (const float*)d_in[61];
    const float* conv0_b    = (const float*)d_in[62];
    const float* conv11_w   = (const float*)d_in[63];
    const float* conv11_b   = (const float*)d_in[64];
    const float* conv12_w   = (const float*)d_in[65];
    const float* conv12_b   = (const float*)d_in[66];
    const float* conv21_w   = (const float*)d_in[67];
    const float* conv21_b   = (const float*)d_in[68];
    const float* conv22_w   = (const float*)d_in[69];
    const float* conv22_b   = (const float*)d_in[70];

    float* ws = (float*)d_ws;
    float* out = (float*)d_out;

    const int S = 512, CH = 256, ROWS = S * CH;      // 131072

    // ---- workspace regions (floats) — reused across phases ----
    float* R_A = ws;                      // 33,554,432  (rows x 256)
    float* R_B = R_A + 33554432;          // 16,777,216  (rows x 128)
    float* R_C = R_B + 16777216;          // 16,777,216
    float* R_D = R_C + 16777216;          // 16,777,216
    float* R_E = R_D + 16777216;          // 16,777,216
    float* R_F = R_E + 16777216;          //  8,388,608  (rows x 64)
    float* R_G = R_F + 8388608;           //  4,194,304  (rows x 32)
    float* R_H = R_G + 4194304;           //  4,194,304
    float* R_S = R_H + 4194304;           //  2,097,152 small

    auto gemm = [&](const float* A, const float* W, const float* bias, float* C,
                    int M, int N, int K, int lda, int ldc, int act) {
        dim3 grid((M + 127) / 128, (N + 63) / 64);
        k_gemm_wmma<<<grid, dim3(256), 0, stream>>>(A, W, bias, C, M, N, K, lda, ldc, act);
    };
    auto elgrid = [](size_t total) { return dim3((unsigned)((total + 255) / 256)); };

    // ================= encode: fast path =================
    float* tG = R_G;   // rows x 32
    float* tH = R_H;   // rows x 32
    k_layernorm<<<elgrid(ROWS), 256, 0, stream>>>(x, tG, in_ln_w, in_ln_b, ROWS, 32);
    gemm(tG, fast_lin_w, fast_lin_b, tH, ROWS, 32, 32, 32, 32, 1);
    k_layernorm<<<elgrid(ROWS), 256, 0, stream>>>(tH, tH, fast_ln_w, fast_ln_b, ROWS, 32);
    // (S,CH,32) -> (CH,S,32)
    k_transpose01<<<elgrid((size_t)ROWS * 32), 256, 0, stream>>>(tH, tG, S, CH, 32);

    // ---- fast Mamba (b=256, L=512, d_inner=64) ----
    float* XZ = R_B;
    gemm(tG, f_in_proj, nullptr, XZ, ROWS, 128, 32, 32, 128, 0);
    float* U = R_F;
    k_dwconv_silu<<<elgrid((size_t)ROWS * 64), 256, 0, stream>>>(XZ, 128, 0, f_conv_w, f_conv_b, U, CH, S, 64);
    float* XD = R_A;                                  // rows x 66
    gemm(U, f_x_proj, nullptr, XD, ROWS, 66, 64, 64, 66, 0);
    float* DT = R_C;                                  // rows x 64
    k_dtproj<<<elgrid((size_t)ROWS * 64), 256, 0, stream>>>(XD, 66, f_dtw, f_dtb, DT, ROWS, 64, 2);
    float* Y = R_D;                                   // rows x 64
    k_scan<<<dim3(CH), dim3(64), 0, stream>>>(U, DT, f_Alog, XD, 66, 2, 34, f_D, Y, S, 64);
    k_gate<<<elgrid((size_t)ROWS * 64), 256, 0, stream>>>(Y, XZ, 128, 64, ROWS, 64);
    float* O1 = tH;                                   // rows x 32 (reuse)
    gemm(Y, f_out_proj, nullptr, O1, ROWS, 32, 64, 64, 32, 0);
    float* XFo = tG;                                  // rows x 32 (reuse)
    gemm(O1, f_wrap_w, f_wrap_b, XFo, ROWS, 32, 32, 32, 32, 1);

    // ---- slow encode ----
    float* E1 = R_C;                                  // rows x 64 (DT dead)
    gemm(XFo, slow_lin_w, slow_lin_b, E1, ROWS, 64, 32, 32, 64, 1);
    k_layernorm<<<elgrid(ROWS), 256, 0, stream>>>(E1, E1, slow_ln_w, slow_ln_b, ROWS, 64);
    float* T2 = R_D;                                  // (CH,S,64) -> (S,CH,64)  (Y dead)
    k_transpose01<<<elgrid((size_t)ROWS * 64), 256, 0, stream>>>(E1, T2, CH, S, 64);

    // ---- slow Mamba (b=512, L=256, d_inner=128) ----
    float* XZ2 = R_A;                                 // rows x 256 (XD dead)
    gemm(T2, s_in_proj, nullptr, XZ2, ROWS, 256, 64, 64, 256, 0);
    float* U2 = R_B;                                  // rows x 128 (XZ dead)
    k_dwconv_silu<<<elgrid((size_t)ROWS * 128), 256, 0, stream>>>(XZ2, 256, 0, s_conv_w, s_conv_b, U2, S, CH, 128);
    float* XD2 = R_C;                                 // rows x 68 (E1 dead)
    gemm(U2, s_x_proj, nullptr, XD2, ROWS, 68, 128, 128, 68, 0);
    float* DT2 = R_D;                                 // rows x 128 (T2 dead)
    k_dtproj<<<elgrid((size_t)ROWS * 128), 256, 0, stream>>>(XD2, 68, s_dtw, s_dtb, DT2, ROWS, 128, 4);
    float* Y2 = R_E;                                  // rows x 128
    k_scan<<<dim3(S), dim3(128), 0, stream>>>(U2, DT2, s_Alog, XD2, 68, 4, 36, s_D, Y2, CH, 128);
    k_gate<<<elgrid((size_t)ROWS * 128), 256, 0, stream>>>(Y2, XZ2, 256, 128, ROWS, 128);
    float* O2 = R_B;                                  // rows x 64 (U2 dead)
    gemm(Y2, s_out_proj, nullptr, O2, ROWS, 64, 128, 128, 64, 0);
    float* XSo = R_F;                                 // rows x 64 (U dead), layout (S,CH,64)
    gemm(O2, s_wrap_w, s_wrap_b, XSo, ROWS, 64, 64, 64, 64, 1);

    // ================= spatial projection =================
    float* T3 = R_D;                                  // (S,CH,64) -> (CH,64,S)
    k_permute120<<<elgrid((size_t)ROWS * 64), 256, 0, stream>>>(XSo, T3, S, CH, 64);
    float* XR = R_S;                                  // (256,64,32)
    gemm(T3, range_w, range_b, XR, CH * 64, 32, 512, 512, 32, 1);
    k_layernorm<<<elgrid(CH * 64), 256, 0, stream>>>(XR, XR, range_ln_w, range_ln_b, CH * 64, 32);
    float* T4 = R_S + 524288;                         // (256,64,32) -> (256,32,64)
    k_permute021<<<elgrid((size_t)CH * 64 * 32), 256, 0, stream>>>(XR, T4, CH, 64, 32);
    float* XSP = R_S + 1048576;                       // (256,32,56) = NCHW C=256,H=32,W=56
    gemm(T4, az_w, az_b, XSP, CH * 32, 56, 64, 64, 56, 1);
    k_layernorm<<<elgrid(CH * 32), 256, 0, stream>>>(XSP, XSP, az_ln_w, az_ln_b, CH * 32, 56);

    // ================= detection decoder =================
    float* DB0 = R_A;
    float* DB1 = R_A + 229376;
    float* DB2 = R_A + 458752;
    const int HW0 = 32 * 56, HW1 = 64 * 112, HW2 = 128 * 224;

    k_conv1x1<<<elgrid((size_t)16 * HW0), 256, 0, stream>>>(XSP, pool_det_w, pool_det_b, DB0, 256, 16, HW0);
    k_conv3x3<<<elgrid((size_t)32 * HW0), 256, 0, stream>>>(DB0, conv1_w, nullptr, DB1, 16, 32, 32, 56, 1);
    k_batchnorm<<<dim3(32), 256, 0, stream>>>(DB1, bn1_g, bn1_b, HW0);
    k_conv3x3<<<elgrid((size_t)16 * HW0), 256, 0, stream>>>(DB1, conv2_w, nullptr, DB0, 32, 16, 32, 56, 1);
    k_batchnorm<<<dim3(16), 256, 0, stream>>>(DB0, bn2_g, bn2_b, HW0);
    k_resize<<<elgrid((size_t)16 * HW1), 256, 0, stream>>>(DB0, DB1, 16, 32, 56, 64, 112);
    k_conv3x3<<<elgrid((size_t)8 * HW1), 256, 0, stream>>>(DB1, conv3_w, nullptr, DB0, 16, 8, 64, 112, 1);
    k_batchnorm<<<dim3(8), 256, 0, stream>>>(DB0, bn3_g, bn3_b, HW1);
    k_resize<<<elgrid((size_t)8 * HW2), 256, 0, stream>>>(DB0, DB1, 8, 64, 112, 128, 224);
    k_conv3x3<<<elgrid((size_t)8 * HW2), 256, 0, stream>>>(DB1, conv4_w, nullptr, DB0, 8, 8, 128, 224, 0);
    k_batchnorm<<<dim3(8), 256, 0, stream>>>(DB0, bn4_g, bn4_b, HW2);
    k_conv3x3<<<elgrid((size_t)1 * HW2), 256, 0, stream>>>(DB0, cls_w, cls_b, out, 8, 1, 128, 224, 2);
    k_conv3x3<<<elgrid((size_t)2 * HW2), 256, 0, stream>>>(DB0, reg_w, reg_b, out + HW2, 8, 2, 128, 224, 0);

    // ================= segmentation decoder =================
    k_conv1x1<<<elgrid((size_t)HW0), 256, 0, stream>>>(XSP, pool_seg_w, pool_seg_b, DB1, 256, 1, HW0);
    k_conv3x3<<<elgrid((size_t)8 * HW0), 256, 0, stream>>>(DB1, conv0_w, conv0_b, DB2, 1, 8, 32, 56, 1);
    k_resize<<<elgrid((size_t)8 * HW1), 256, 0, stream>>>(DB2, DB1, 8, 32, 56, 64, 112);
    k_conv3x3<<<elgrid((size_t)8 * HW1), 256, 0, stream>>>(DB1, conv11_w, conv11_b, DB2, 8, 8, 64, 112, 1);
    k_conv3x3<<<elgrid((size_t)4 * HW1), 256, 0, stream>>>(DB2, conv12_w, conv12_b, DB1, 8, 4, 64, 112, 1);
    k_resize<<<elgrid((size_t)4 * HW2), 256, 0, stream>>>(DB1, DB2, 4, 64, 112, 128, 224);
    k_conv3x3<<<elgrid((size_t)4 * HW2), 256, 0, stream>>>(DB2, conv21_w, conv21_b, DB1, 4, 4, 128, 224, 1);
    k_conv3x3<<<elgrid((size_t)HW2), 256, 0, stream>>>(DB1, conv22_w, conv22_b, out + 3 * HW2, 4, 1, 128, 224, 0);
}
